// Aggregator_27444841021984
// MI455X (gfx1250) — compile-verified, matrix operand order checked by reading, not run on hardware
//
#include <hip/hip_runtime.h>
#include <hip/hip_bf16.h>

typedef __attribute__((ext_vector_type(16))) _Float16 v16h;
typedef __attribute__((ext_vector_type(8)))  float    v8f;

#define N_ENT   200000
#define N_ITEM  50000
#define N_USER  100000
#define DIM     64
#define N_EDGE  2000000
#define HIST    20
#define NNZ_I   1600000
#define NNZ_U   3200000

// ---- workspace layout (floats) ----
#define WS_KG_SUM    0                         // N_ENT*DIM   (becomes kg_agg in place)
#define WS_CNT       (WS_KG_SUM + N_ENT*DIM)   // N_ENT
#define WS_ITEM_NB   (WS_CNT + N_ENT)          // N_ITEM*DIM
#define WS_USER_CO   (WS_ITEM_NB + N_ITEM*DIM) // N_USER*DIM
#define WS_SCORE     (WS_USER_CO + N_USER*DIM) // N_USER*HIST
#define WS_UIA       (WS_SCORE + N_USER*HIST)  // N_USER*DIM
#define WS_ZERO_N    WS_SCORE                  // region that needs zeroing (atomic targets)

// ------------------------------------------------------------------
// WMMA fragment loader: row-major f32 source row (ld = 64) -> f16 frag.
// 16-bit A/B layout (ISA 7.12.2): element e of lane (h2 = lane>>4):
//   K = (e>>3)*16 + h2*8 + (e&7)
// ------------------------------------------------------------------
__device__ __forceinline__ v16h load_frag(const float* row, int kk, int h2) {
  v16h f;
#pragma unroll
  for (int e = 0; e < 16; ++e) {
    int k = kk + ((e >> 3) << 4) + h2 * 8 + (e & 7);
    f[e] = (_Float16)row[k];
  }
  return f;
}

// ------------------------------------------------------------------
// zero scratch (atomic accumulation targets)
// ------------------------------------------------------------------
__global__ void zero_ws_kernel(float* ws, int n) {
  int i = blockIdx.x * blockDim.x + threadIdx.x;
  if (i < n) ws[i] = 0.0f;
}

// ------------------------------------------------------------------
// KG edge scatter: 16 threads/edge, 4 floats each (float4 row reads)
// ------------------------------------------------------------------
__global__ void edge_scatter_kernel(const float* __restrict__ ent,
                                    const float* __restrict__ weight,
                                    const int* __restrict__ eidx,
                                    const int* __restrict__ etype,
                                    float* __restrict__ kg_sum,
                                    float* __restrict__ cnt) {
  int gid = blockIdx.x * blockDim.x + threadIdx.x;
  int e = gid >> 4;
  if (e >= N_EDGE) return;
  int q = gid & 15;
  int head = eidx[e];
  int tail = eidx[N_EDGE + e];
  int rel  = etype[e] - 1;
  const float4 ev = *reinterpret_cast<const float4*>(ent    + (size_t)tail * DIM + q * 4);
  const float4 wv = *reinterpret_cast<const float4*>(weight + (size_t)rel  * DIM + q * 4);
  float* dst = kg_sum + (size_t)head * DIM + q * 4;
  atomicAdd(dst + 0, ev.x * wv.x);
  atomicAdd(dst + 1, ev.y * wv.y);
  atomicAdd(dst + 2, ev.z * wv.z);
  atomicAdd(dst + 3, ev.w * wv.w);
  if (q == 0) atomicAdd(cnt + head, 1.0f);
}

// ------------------------------------------------------------------
// generic sparse scatter: out[row] += val * src[col]   (16 thr / nnz)
// ------------------------------------------------------------------
__global__ void spmm_scatter_kernel(const float* __restrict__ src,
                                    const int* __restrict__ rows,
                                    const int* __restrict__ cols,
                                    const float* __restrict__ vals,
                                    float* __restrict__ out, int nnz) {
  int gid = blockIdx.x * blockDim.x + threadIdx.x;
  int e = gid >> 4;
  if (e >= nnz) return;
  int q = gid & 15;
  int r = rows[e], c = cols[e];
  float v = vals[e];
  const float4 sv = *reinterpret_cast<const float4*>(src + (size_t)c * DIM + q * 4);
  float* dst = out + (size_t)r * DIM + q * 4;
  atomicAdd(dst + 0, v * sv.x);
  atomicAdd(dst + 1, v * sv.y);
  atomicAdd(dst + 2, v * sv.z);
  atomicAdd(dst + 3, v * sv.w);
}

// ------------------------------------------------------------------
// kg_agg = kg_sum / max(cnt,1)   (in place); entities >= N_ITEM go
// straight to the output.
// ------------------------------------------------------------------
__global__ void kg_finalize_kernel(float* __restrict__ kg_sum,
                                   const float* __restrict__ cnt,
                                   float* __restrict__ out_ent) {
  int i = blockIdx.x * blockDim.x + threadIdx.x;   // over N_ENT*DIM
  if (i >= N_ENT * DIM) return;
  int ent = i >> 6;
  float c = cnt[ent];
  float v = kg_sum[i] / fmaxf(c, 1.0f);
  kg_sum[i] = v;
  if (ent >= N_ITEM) out_ent[i] = v;
}

// ------------------------------------------------------------------
// Gated dual linear (WMMA):
//   s   = sigmoid(X1@W1^T + b1 + X2@W2^T + b2)
//   out = s*X1 + (1-s)*X2
// block = 128 threads (4 waves); block handles 16 rows; wave w owns
// column tile [16w,16w+16). M must be a multiple of 16.
// ------------------------------------------------------------------
__global__ __launch_bounds__(128)
void gated_linear_kernel(const float* __restrict__ X1,
                         const float* __restrict__ X2,
                         const float* __restrict__ W1,
                         const float* __restrict__ b1,
                         const float* __restrict__ W2,
                         const float* __restrict__ b2,
                         float* __restrict__ out) {
  const int lane = threadIdx.x & 31;
  const int wave = threadIdx.x >> 5;
  const int h2   = lane >> 4;
  const int m    = lane & 15;
  const int row0 = blockIdx.x * 16;
  const int col0 = wave * 16;

  v8f acc = {};
#pragma unroll
  for (int kk = 0; kk < DIM; kk += 32) {
    v16h a1 = load_frag(X1 + (size_t)(row0 + m) * DIM, kk, h2);
    v16h w1 = load_frag(W1 + (size_t)(col0 + m) * DIM, kk, h2);   // B[k][n] = W[n][k]
    acc = __builtin_amdgcn_wmma_f32_16x16x32_f16(false, a1, false, w1,
                                                 (short)0, acc, false, false);
    v16h a2 = load_frag(X2 + (size_t)(row0 + m) * DIM, kk, h2);
    v16h w2 = load_frag(W2 + (size_t)(col0 + m) * DIM, kk, h2);
    acc = __builtin_amdgcn_wmma_f32_16x16x32_f16(false, a2, false, w2,
                                                 (short)0, acc, false, false);
  }
  const int n = col0 + m;                 // output column for this lane
  const float bias = b1[n] + b2[n];
#pragma unroll
  for (int r = 0; r < 8; ++r) {
    int row = row0 + r + 8 * h2;
    float pre = acc[r] + bias;
    float s = 1.0f / (1.0f + __expf(-pre));
    float x1 = X1[(size_t)row * DIM + n];
    float x2 = X2[(size_t)row * DIM + n];
    out[(size_t)row * DIM + n] = s * x1 + (1.0f - s) * x2;
  }
}

// ------------------------------------------------------------------
// Attention scores (WMMA):
//   score[u,h] = sum_d tanh(item_emb[idx[u,h]] @ W_ua^T + b_ua)[d] * user_emb[u,d]
// block = 128 threads / 4 waves; 16 flattened (u,h) rows per block.
// ------------------------------------------------------------------
__global__ __launch_bounds__(128)
void attn_score_kernel(const float* __restrict__ ent,       // item_emb = entity_emb[:N_ITEM]
                       const float* __restrict__ user_emb,
                       const int* __restrict__ sample,      // N_USER*HIST
                       const float* __restrict__ W_ua,
                       const float* __restrict__ b_ua,
                       float* __restrict__ score) {
  __shared__ float smem[4][16][16];
  const int lane = threadIdx.x & 31;
  const int wave = threadIdx.x >> 5;
  const int h2   = lane >> 4;
  const int m    = lane & 15;
  const int row0 = blockIdx.x * 16;     // flattened (u,h) row
  const int col0 = wave * 16;

  // A rows are gathered item embeddings
  const int my_item = sample[row0 + m];
  const float* arow = ent + (size_t)my_item * DIM;

  v8f acc = {};
#pragma unroll
  for (int kk = 0; kk < DIM; kk += 32) {
    v16h a = load_frag(arow, kk, h2);
    v16h b = load_frag(W_ua + (size_t)(col0 + m) * DIM, kk, h2);
    acc = __builtin_amdgcn_wmma_f32_16x16x32_f16(false, a, false, b,
                                                 (short)0, acc, false, false);
  }
  const int n = col0 + m;
  const float bias = b_ua[n];
#pragma unroll
  for (int r = 0; r < 8; ++r) {
    int row = row0 + r + 8 * h2;        // flattened (u,h)
    int u = row / HIST;
    float att = tanhf(acc[r] + bias);
    smem[wave][r + 8 * h2][m] = att * user_emb[(size_t)u * DIM + n];
  }
  __syncthreads();
  if (threadIdx.x < 16) {
    float s = 0.0f;
#pragma unroll
    for (int w = 0; w < 4; ++w)
#pragma unroll
      for (int c = 0; c < 16; ++c) s += smem[w][threadIdx.x][c];
    score[row0 + threadIdx.x] = s;
  }
}

// ------------------------------------------------------------------
// Per-user softmax over HIST + weighted sum of gathered item rows.
// block = 64 threads = one user; thread = one dim.
// ------------------------------------------------------------------
__global__ __launch_bounds__(64)
void attn_agg_kernel(const float* __restrict__ ent,
                     const int* __restrict__ sample,
                     const float* __restrict__ score,
                     float* __restrict__ uia) {
  const int u = blockIdx.x;
  const int d = threadIdx.x;
  const float* sc = score + (size_t)u * HIST;
  float mx = -1e30f;
#pragma unroll
  for (int h = 0; h < HIST; ++h) mx = fmaxf(mx, sc[h]);
  float p[HIST];
  float sum = 0.0f;
#pragma unroll
  for (int h = 0; h < HIST; ++h) { p[h] = __expf(sc[h] - mx); sum += p[h]; }
  float inv = 1.0f / sum;
  float accv = 0.0f;
#pragma unroll
  for (int h = 0; h < HIST; ++h) {
    int it = sample[(size_t)u * HIST + h];
    accv += p[h] * inv * ent[(size_t)it * DIM + d];
  }
  uia[(size_t)u * DIM + d] = accv;
}

// ------------------------------------------------------------------
// reg = sum of squares of the 5 weight matrices + 5 biases
// ------------------------------------------------------------------
__global__ __launch_bounds__(256)
void reg_kernel(const float* W_ua, const float* b_ua,
                const float* W_ikg, const float* b_ikg,
                const float* W_ing, const float* b_ing,
                const float* W_uig, const float* b_uig,
                const float* W_ung, const float* b_ung,
                float* out_reg) {
  __shared__ float red[256];
  const float* mats[5] = {W_ua, W_ikg, W_ing, W_uig, W_ung};
  const float* vecs[5] = {b_ua, b_ikg, b_ing, b_uig, b_ung};
  float s = 0.0f;
  for (int a = 0; a < 5; ++a) {
    for (int i = threadIdx.x; i < DIM * DIM; i += 256) { float v = mats[a][i]; s += v * v; }
    for (int i = threadIdx.x; i < DIM;       i += 256) { float v = vecs[a][i]; s += v * v; }
  }
  red[threadIdx.x] = s;
  __syncthreads();
  for (int st = 128; st > 0; st >>= 1) {
    if (threadIdx.x < st) red[threadIdx.x] += red[threadIdx.x + st];
    __syncthreads();
  }
  if (threadIdx.x == 0) *out_reg = red[0];
}

// ------------------------------------------------------------------
extern "C" void kernel_launch(void* const* d_in, const int* in_sizes, int n_in,
                              void* d_out, int out_size, void* d_ws, size_t ws_size,
                              hipStream_t stream) {
  (void)in_sizes; (void)n_in; (void)out_size; (void)ws_size;
  const float* ent      = (const float*)d_in[0];
  const float* user_emb = (const float*)d_in[1];
  const int*   eidx     = (const int*)  d_in[2];
  const int*   etype    = (const int*)  d_in[3];
  const float* weight   = (const float*)d_in[4];
  const int*   un_rows  = (const int*)  d_in[6];
  const int*   un_cols  = (const int*)  d_in[7];
  const float* un_vals  = (const float*)d_in[8];
  const int*   in_rows  = (const int*)  d_in[9];
  const int*   in_cols  = (const int*)  d_in[10];
  const float* in_vals  = (const float*)d_in[11];
  const int*   sample   = (const int*)  d_in[12];
  const float* W_ua  = (const float*)d_in[13];
  const float* b_ua  = (const float*)d_in[14];
  const float* W_ikg = (const float*)d_in[15];
  const float* b_ikg = (const float*)d_in[16];
  const float* W_ing = (const float*)d_in[17];
  const float* b_ing = (const float*)d_in[18];
  const float* W_uig = (const float*)d_in[19];
  const float* b_uig = (const float*)d_in[20];
  const float* W_ung = (const float*)d_in[21];
  const float* b_ung = (const float*)d_in[22];

  float* ws       = (float*)d_ws;
  float* kg_sum   = ws + WS_KG_SUM;   // becomes kg_agg
  float* cnt      = ws + WS_CNT;
  float* item_nb  = ws + WS_ITEM_NB;
  float* user_co  = ws + WS_USER_CO;
  float* score    = ws + WS_SCORE;
  float* uia      = ws + WS_UIA;

  float* out_ent  = (float*)d_out;                       // N_ENT*DIM
  float* out_user = out_ent + (size_t)N_ENT * DIM;       // N_USER*DIM
  float* out_reg  = out_user + (size_t)N_USER * DIM;     // 1

  // 1) zero atomic targets
  zero_ws_kernel<<<(WS_ZERO_N + 255) / 256, 256, 0, stream>>>(ws, WS_ZERO_N);

  // 2) KG edge scatter -> kg_sum, cnt
  edge_scatter_kernel<<<(N_EDGE * 16) / 256, 256, 0, stream>>>(
      ent, weight, eidx, etype, kg_sum, cnt);

  // 3) item-item neighbor sum
  spmm_scatter_kernel<<<(NNZ_I * 16) / 256, 256, 0, stream>>>(
      ent, in_rows, in_cols, in_vals, item_nb, NNZ_I);

  // 4) user-user collab sum
  spmm_scatter_kernel<<<(NNZ_U * 16) / 256, 256, 0, stream>>>(
      user_emb, un_rows, un_cols, un_vals, user_co, NNZ_U);

  // 5) kg_agg = kg_sum / max(cnt,1); write entity output for ent >= N_ITEM
  kg_finalize_kernel<<<(N_ENT * DIM) / 256, 256, 0, stream>>>(kg_sum, cnt, out_ent);

  // 6) item gate (WMMA): entity output rows < N_ITEM
  gated_linear_kernel<<<N_ITEM / 16, 128, 0, stream>>>(
      kg_sum, item_nb, W_ikg, b_ikg, W_ing, b_ing, out_ent);

  // 7) attention scores (WMMA over 2M gathered rows)
  attn_score_kernel<<<(N_USER * HIST) / 16, 128, 0, stream>>>(
      ent, user_emb, sample, W_ua, b_ua, score);

  // 8) softmax + weighted item gather -> user_item_agg
  attn_agg_kernel<<<N_USER, 64, 0, stream>>>(ent, sample, score, uia);

  // 9) user gate (WMMA): user output
  gated_linear_kernel<<<N_USER / 16, 128, 0, stream>>>(
      uia, user_co, W_uig, b_uig, W_ung, b_ung, out_user);

  // 10) reg scalar
  reg_kernel<<<1, 256, 0, stream>>>(W_ua, b_ua, W_ikg, b_ikg, W_ing, b_ing,
                                    W_uig, b_uig, W_ung, b_ung, out_reg);
}
